// MultiHeadAttention_45268955299870
// MI455X (gfx1250) — matrix operand order repaired and checked by reading, not compiled, not run on hardware
//
#include <hip/hip_runtime.h>
#include <hip/hip_bf16.h>
#include <stdint.h>

// ---------------------------------------------------------------------------
// MI455X (gfx1250) multi-head attention, f16 WMMA with f32 accumulate.
// B=4 T=2048 H=8 D=64.  Matmuls -> v_wmma_f32_16x16x32_f16; K tiles staged
// to LDS with global_load_async_to_lds_b128 (ASYNCcnt, double-buffered);
// softmax fully register-resident with wave32 butterfly reductions.
// ---------------------------------------------------------------------------

typedef _Float16 f16;
typedef __attribute__((ext_vector_type(16))) _Float16 v16h;
typedef __attribute__((ext_vector_type(8)))  float    v8f;
typedef __attribute__((ext_vector_type(8)))  uint32_t v8u;
typedef uint32_t __attribute__((may_alias)) u32a;

#define WMMA_F32_F16(a, b, c) \
  __builtin_amdgcn_wmma_f32_16x16x32_f16(false, (a), false, (b), (short)0, (c), false, false)

// A-fragment (16x32 f16, M x K). Lane holds row M=lane&15;
// dword pair p holds K = 2*p + 16*(lane>>4).
__device__ __forceinline__ v16h load_a_frag(const f16* base, int stride, int lane) {
  const f16* row = base + (lane & 15) * stride;
  const int khi = (lane >> 4) << 4;
  v8u u;
#pragma unroll
  for (int p = 0; p < 8; ++p) {
    int k = (p << 1) + khi;
    u[p] = *(const u32a*)(row + k);
  }
  return __builtin_bit_cast(v16h, u);
}

// B-fragment (32x16 f16, K x N) from an N-major source: B[k][n] = Src[n][k].
// Lane holds column N=lane&15 (one source row); element e holds K = e + 16*(lane>>4).
__device__ __forceinline__ v16h load_b_frag(const f16* base, int stride, int lane) {
  const f16* row = base + (lane & 15) * stride + ((lane >> 4) << 4);
  v8u u;
#pragma unroll
  for (int v = 0; v < 8; ++v) u[v] = *(const u32a*)(row + 2 * v);
  return __builtin_bit_cast(v16h, u);
}

// Async copy of one 64-f16 (128B) K row into LDS: 8 x b128 per lane.
// VDST = wave-relative LDS byte address, VADDR = 64-bit global address.
__device__ __forceinline__ void stage_k_row(f16* lds_row, const f16* grow) {
  uint32_t ldsa = (uint32_t)(uintptr_t)lds_row;
#pragma unroll
  for (int q = 0; q < 8; ++q) {
    asm volatile("global_load_async_to_lds_b128 %0, %1, off"
                 :: "v"(ldsa + 16u * q), "v"(grow + q * 8)
                 : "memory");
  }
}
__device__ __forceinline__ void wait_async_le8() {
  asm volatile("s_wait_asynccnt 0x8" ::: "memory");
}
__device__ __forceinline__ void wait_async_0() {
  asm volatile("s_wait_asynccnt 0x0" ::: "memory");
}

// ---------------------------------------------------------------------------
__global__ void cvt_f16_kernel(const float* __restrict__ in, f16* __restrict__ out, int n) {
  int i = blockIdx.x * blockDim.x + threadIdx.x;
  if (i < n) out[i] = (f16)in[i];
}

// values [B,T,H,64] (f32) -> vt [B,H,64,T] (f16): P*V B-fragments contiguous in t.
__global__ void vtrans_f16_kernel(const float* __restrict__ v, f16* __restrict__ vt) {
  int i = blockIdx.x * blockDim.x + threadIdx.x;   // out index over [4][8][64][2048]
  if (i >= 4 * 8 * 64 * 2048) return;
  int t = i & 2047;
  int d = (i >> 11) & 63;
  int h = (i >> 17) & 7;
  int b = i >> 20;
  vt[i] = (f16)v[((b * 2048 + t) * 8 + h) * 64 + d];
}

// ---------------------------------------------------------------------------
// Q = X[8192,512] @ Wq[512,512]^T -> f16 [8192,512].
// One wave per 16x64 output stripe (4 tiles share each A fragment).
__global__ void __launch_bounds__(256) qproj_kernel(const f16* __restrict__ X,
                                                    const f16* __restrict__ W,
                                                    f16* __restrict__ Q) {
  int warp = threadIdx.x >> 5, lane = threadIdx.x & 31;
  int w = blockIdx.x * 8 + warp;       // 0..4095
  int R0 = (w >> 3) << 4;              // 512 row tiles
  int C0 = (w & 7) << 6;               // 8 groups of 64 cols
  v8f acc[4];
#pragma unroll
  for (int j = 0; j < 4; ++j)
#pragma unroll
    for (int r = 0; r < 8; ++r) acc[j][r] = 0.0f;

  for (int ch = 0; ch < 16; ++ch) {
    v16h a = load_a_frag(X + R0 * 512 + ch * 32, 512, lane);
#pragma unroll
    for (int j = 0; j < 4; ++j) {
      v16h b = load_b_frag(W + (C0 + j * 16) * 512 + ch * 32, 512, lane);
      acc[j] = WMMA_F32_F16(a, b, acc[j]);
    }
  }
  int lo = lane & 15, mbase = R0 + ((lane >> 4) << 3);
#pragma unroll
  for (int j = 0; j < 4; ++j) {
    int n = C0 + j * 16 + lo;
#pragma unroll
    for (int r = 0; r < 8; ++r) Q[(mbase + r) * 512 + n] = (f16)acc[j][r];
  }
}

// ---------------------------------------------------------------------------
// Flash attention: one wave per (b, h, 16-row q tile); 4 waves/block.
// K chunks (32x64 f16) async-staged to LDS, double-buffered on ASYNCcnt.
// Softmax in registers: row stats via 4-step shfl_xor butterflies over the
// 16 lanes sharing a C-fragment row.  Diagonal masked to -1e9 pre-scale.
#define ATTN_WARPS 4
__global__ void __launch_bounds__(128) attn_kernel(const f16* __restrict__ Qh,
                                                   const f16* __restrict__ Kh,
                                                   const f16* __restrict__ Vt,
                                                   f16* __restrict__ O16) {
  __shared__ f16 Kstage[ATTN_WARPS][2][32][64];  // 32 KB
  __shared__ f16 Plds[ATTN_WARPS][512];          // 4 KB (C-frag -> A-frag relayout)

  const int warp = threadIdx.x >> 5, lane = threadIdx.x & 31;
  const int w = blockIdx.x * ATTN_WARPS + warp;  // 0..4095
  const int qt = w & 127, h = (w >> 7) & 7, b = w >> 10;
  const int q0 = qt << 4;
  const int hi = lane >> 4, lo = lane & 15;

  const f16* qbase = Qh + (b * 2048 + q0) * 512 + h * 64;
  v16h aQ0 = load_a_frag(qbase,      512, lane);   // d = 0..31
  v16h aQ1 = load_a_frag(qbase + 32, 512, lane);   // d = 32..63

  const f16* krows = Kh + (size_t)(b * 2048) * 512 + h * 64;  // + (kc+row)*512
  const f16* vbase = Vt + (size_t)((b * 8 + h) * 64) * 2048;  // rows = dv, cols = t

  // prime buffer 0 with chunk 0 (each lane stages one K row)
  stage_k_row(&Kstage[warp][0][lane][0], krows + (size_t)lane * 512);

  v8f O[4];
#pragma unroll
  for (int dt = 0; dt < 4; ++dt)
#pragma unroll
    for (int r = 0; r < 8; ++r) O[dt][r] = 0.0f;
  float mst[8], lst[8];
#pragma unroll
  for (int r = 0; r < 8; ++r) { mst[r] = -3.0e38f; lst[r] = 0.0f; }

  for (int kc = 0; kc < 2048; kc += 32) {
    const int buf = (kc >> 5) & 1;
    const bool more = (kc + 32) < 2048;
    if (more)
      stage_k_row(&Kstage[warp][buf ^ 1][lane][0], krows + (size_t)(kc + 32 + lane) * 512);
    if (more) wait_async_le8(); else wait_async_0();

    // ---- S = Q K^T : two 16x16 n-tiles from the staged chunk
    v8f s0 = {}, s1 = {};
    {
      const f16* kb0 = &Kstage[warp][buf][0][0];
      const f16* kb1 = &Kstage[warp][buf][16][0];
      s0 = WMMA_F32_F16(aQ0, load_b_frag(kb0,      64, lane), s0);
      s0 = WMMA_F32_F16(aQ1, load_b_frag(kb0 + 32, 64, lane), s0);
      s1 = WMMA_F32_F16(aQ0, load_b_frag(kb1,      64, lane), s1);
      s1 = WMMA_F32_F16(aQ1, load_b_frag(kb1 + 32, 64, lane), s1);
    }

    // ---- mask (pre-scale, as reference) + scale, in C-fragment registers
#pragma unroll
    for (int r = 0; r < 8; ++r) {
      const int mrow = q0 + r + hi * 8;           // global q index of this element
      float a0 = s0[r]; if (kc + lo == mrow)      a0 = -1.0e9f;
      float a1 = s1[r]; if (kc + 16 + lo == mrow) a1 = -1.0e9f;
      s0[r] = a0 * 0.125f;
      s1[r] = a1 * 0.125f;
    }

    // ---- online softmax, row stats via butterflies over the 16-lane row group
    float corr[8], mnew[8];
#pragma unroll
    for (int r = 0; r < 8; ++r) {
      float mx = fmaxf(s0[r], s1[r]);
      mx = fmaxf(mx, __shfl_xor(mx, 1));
      mx = fmaxf(mx, __shfl_xor(mx, 2));
      mx = fmaxf(mx, __shfl_xor(mx, 4));
      mx = fmaxf(mx, __shfl_xor(mx, 8));
      float mn = fmaxf(mst[r], mx);
      corr[r] = __expf(mst[r] - mn);
      mnew[r] = mn;
      mst[r] = mn;
    }
#pragma unroll
    for (int r = 0; r < 8; ++r) {
      float p0 = __expf(s0[r] - mnew[r]);
      float p1 = __expf(s1[r] - mnew[r]);
      Plds[warp][(r + hi * 8) * 32 + lo]      = (f16)p0;
      Plds[warp][(r + hi * 8) * 32 + 16 + lo] = (f16)p1;
      float rs = p0 + p1;
      rs += __shfl_xor(rs, 1);
      rs += __shfl_xor(rs, 2);
      rs += __shfl_xor(rs, 4);
      rs += __shfl_xor(rs, 8);
      lst[r] = lst[r] * corr[r] + rs;
    }

    // ---- O = O*corr + P V  (A-frag relayout of P through warp-private LDS;
    //      same-wave DS ops are in-order, no barrier needed)
    v16h aP = load_a_frag(&Plds[warp][0], 32, lane);
#pragma unroll
    for (int dt = 0; dt < 4; ++dt) {
#pragma unroll
      for (int r = 0; r < 8; ++r) O[dt][r] *= corr[r];
      const f16* vrow = vbase + (size_t)(dt * 16) * 2048 + kc;
      v16h bv = load_b_frag(vrow, 2048, lane);
      if (kc + 160 < 2048)
        __builtin_prefetch(vrow + lo * 2048 + (hi << 4) + 128, 0, 0);  // global_prefetch_b8
      O[dt] = WMMA_F32_F16(aP, bv, O[dt]);
    }
  }

  // ---- finalize: divide by l (stats uniform across each row group), write f16
#pragma unroll
  for (int r = 0; r < 8; ++r) lst[r] = 1.0f / lst[r];
#pragma unroll
  for (int dt = 0; dt < 4; ++dt)
#pragma unroll
    for (int r = 0; r < 8; ++r) {
      int t = q0 + r + hi * 8;
      O16[((b * 2048 + t) * 8 + h) * 64 + dt * 16 + lo] = (f16)(O[dt][r] * lst[r]);
    }
}

// ---------------------------------------------------------------------------
// out[r][e] = sum_d X[r][d] * Wfc[e][d] + bias[e], rows = B*T*H = 65536, f32 out.
__global__ void __launch_bounds__(256) fc_kernel(const f16* __restrict__ X,
                                                 const f16* __restrict__ Wf,
                                                 const float* __restrict__ bias,
                                                 float* __restrict__ out) {
  int warp = threadIdx.x >> 5, lane = threadIdx.x & 31;
  int w = blockIdx.x * 8 + warp;   // 0..4095
  int r0 = w << 4;
  const int hi = lane >> 4, lo = lane & 15;
  v16h a0 = load_a_frag(X + r0 * 64,      64, lane);
  v16h a1 = load_a_frag(X + r0 * 64 + 32, 64, lane);
#pragma unroll
  for (int et = 0; et < 4; ++et) {
    v16h b0 = load_b_frag(Wf + (et * 16) * 64,      64, lane);
    v16h b1 = load_b_frag(Wf + (et * 16) * 64 + 32, 64, lane);
    v8f c = {};
    c = WMMA_F32_F16(a0, b0, c);
    c = WMMA_F32_F16(a1, b1, c);
    float bb = bias[et * 16 + lo];
#pragma unroll
    for (int r = 0; r < 8; ++r)
      out[(r0 + r + hi * 8) * 64 + et * 16 + lo] = c[r] + bb;
  }
}

// ---------------------------------------------------------------------------
extern "C" void kernel_launch(void* const* d_in, const int* in_sizes, int n_in,
                              void* d_out, int out_size, void* d_ws, size_t ws_size,
                              hipStream_t stream) {
  const float* values  = (const float*)d_in[0];  // [4,2048,8,64]
  const float* keys    = (const float*)d_in[1];  // [4,2048,8,64]
  const float* queries = (const float*)d_in[2];  // [4,2048,8,64] -> [8192,512]
  const float* Wq      = (const float*)d_in[3];  // [512,512]
  const float* Wfc     = (const float*)d_in[4];  // [64,64]
  const float* bfc     = (const float*)d_in[5];  // [64]
  float* out = (float*)d_out;                    // [4,2048,8,64] f32

  char* ws = (char*)d_ws;
  size_t off = 0;
  auto alloc = [&](size_t bytes) -> void* {
    void* p = ws + off;
    off += (bytes + 255) & ~(size_t)255;
    return p;
  };
  f16* qin16 = (f16*)alloc((size_t)8192 * 512 * 2);  // queries f16
  f16* k16   = (f16*)alloc((size_t)8192 * 512 * 2);  // keys f16
  f16* vt16  = (f16*)alloc((size_t)8192 * 512 * 2);  // V^T [B,H,64,T] f16
  f16* wq16  = (f16*)alloc((size_t)512 * 512 * 2);
  f16* wfc16 = (f16*)alloc((size_t)64 * 64 * 2);
  f16* q16   = (f16*)alloc((size_t)8192 * 512 * 2);  // projected Q f16
  f16* at16  = (f16*)alloc((size_t)8192 * 512 * 2);  // attention output f16
  (void)ws_size; (void)in_sizes; (void)n_in; (void)out_size;

  cvt_f16_kernel<<<(4194304 + 255) / 256, 256, 0, stream>>>(queries, qin16, 4194304);
  cvt_f16_kernel<<<(4194304 + 255) / 256, 256, 0, stream>>>(keys, k16, 4194304);
  cvt_f16_kernel<<<(262144 + 255) / 256, 256, 0, stream>>>(Wq, wq16, 262144);
  cvt_f16_kernel<<<16, 256, 0, stream>>>(Wfc, wfc16, 4096);
  vtrans_f16_kernel<<<16384, 256, 0, stream>>>(values, vt16);

  qproj_kernel<<<512, 256, 0, stream>>>(qin16, wq16, q16);       // 4096 waves
  attn_kernel<<<1024, 128, 0, stream>>>(q16, k16, vt16, at16);   // 4096 waves
  fc_kernel<<<512, 256, 0, stream>>>(at16, wfc16, bfc, out);     // 4096 waves
}
